// MeshConvBlock_18494129176647
// MI455X (gfx1250) — matrix operand order repaired and checked by reading, not compiled, not run on hardware
//
#include <hip/hip_runtime.h>
#include <hip/hip_bf16.h>
#include <math.h>

typedef __attribute__((ext_vector_type(16))) __bf16 bf16x16;
typedef __attribute__((ext_vector_type(8)))  float  f32x8;

#define DIM 128

#if __has_builtin(__builtin_amdgcn_tensor_load_to_lds)
#define HAS_TDM 1
#else
#define HAS_TDM 0
#endif

// ---------------------------------------------------------------------------
// TDM 1-D bulk copy global -> LDS (CDNA5 ISA D# layout).
// data_size=3 (8B elements), 1-D tile: tensor_dim0 = tile_dim0 = nqwords.
// Issued once per wave; completion via s_wait_tensorcnt.
// 6-arg builtin variant: (g0, g1, g2, g3, g4, cpol).
// ---------------------------------------------------------------------------
#if HAS_TDM
typedef __attribute__((ext_vector_type(4))) unsigned int u32x4;
typedef __attribute__((ext_vector_type(8))) int i32x8;
typedef __attribute__((ext_vector_type(4))) int i32x4;

__device__ __forceinline__ void tdm_load_1d(void* lds_ptr, const void* gptr, unsigned bytes) {
    unsigned lds_off = (unsigned)(uintptr_t)lds_ptr;         // flat LDS addr[31:0] = LDS offset
    unsigned long long ga = (unsigned long long)(uintptr_t)gptr;
    unsigned nq = bytes >> 3;                                // 8-byte elements (<= 65535)
    u32x4 g0;
    g0[0] = 1u;                                              // count=1, user descriptor
    g0[1] = lds_off;                                         // lds_addr
    g0[2] = (unsigned)ga;                                    // global_addr[31:0]
    g0[3] = (unsigned)((ga >> 32) & 0x01FFFFFFu) | (2u << 30); // addr[56:32] | type=2
    i32x8 g1;
    g1[0] = (3 << 16);                                       // wg_mask=0, data_size=3 (8B)
    g1[1] = (int)((nq & 0xFFFFu) << 16);                     // tensor_dim0[15:0]
    g1[2] = (int)(((nq >> 16) & 0xFFFFu) | (1u << 16));      // dim0[31:16] | tensor_dim1=1
    g1[3] = (int)((nq & 0xFFFFu) << 16);                     // dim1 hi=0 | tile_dim0=nq
    g1[4] = 0;                                               // tile_dim1=0 (1-D), tile_dim2=0
    g1[5] = (int)nq;                                         // tensor_dim0_stride lo
    g1[6] = 0;
    g1[7] = 0;
    i32x4 z4 = {0, 0, 0, 0};
    i32x8 z8 = {0, 0, 0, 0, 0, 0, 0, 0};
    __builtin_amdgcn_tensor_load_to_lds(g0, g1, z4, z4, z8, 0);
}
#endif

// Stage `bytes` from global into LDS; wave0 uses TDM, else cooperative copy.
// Must be called by ALL threads of the block; ends with a block barrier.
__device__ __forceinline__ void stage_to_lds(void* lds_ptr, const void* gptr,
                                             unsigned bytes, int tid) {
#if HAS_TDM
    if (tid < 32) {
        tdm_load_1d(lds_ptr, gptr, bytes);
        __builtin_amdgcn_s_wait_tensorcnt(0);
    }
#else
    const uint4* s = (const uint4*)gptr;
    uint4* d = (uint4*)lds_ptr;
    for (unsigned i = tid; i < (bytes >> 4); i += 256) d[i] = s[i];
#endif
    __syncthreads();
}

// ---------------------------------------------------------------------------
// Fragment loaders (CDNA5 wave32 WMMA 16x16x32 bf16 layouts, CDNA5 ISA)
// ---------------------------------------------------------------------------
__device__ __forceinline__ bf16x16 load_a_frag_lds(const __bf16* a, int lane, int ks) {
    int m  = lane & 15;
    int kb = ks * 32 + ((lane & 16) ? 8 : 0);
    const uint4* p = (const uint4*)(a + m * DIM + kb);   // 16B aligned
    union { uint4 u[2]; bf16x16 v; } t;
    t.u[0] = p[0];   // K = kb .. kb+7
    t.u[1] = p[2];   // K = kb+16 .. kb+23
    return t.v;
}

// B fragments pre-swizzled: frag (nt,ks), lane owns 16 contiguous bf16 (32B).
__device__ __forceinline__ bf16x16 load_b_frag(const __bf16* w, int nt, int ks, int lane) {
    return *(const bf16x16*)(w + ((size_t)((nt * 4 + ks) * 32 + lane)) * 16);
}

// ---------------------------------------------------------------------------
// Kernel 0: swizzle weights into B-fragment order (bf16).
//   wA = W1[0:128]-W1[128:256], wB = W1[128:256], w2 = W2   (each 128x128)
// ---------------------------------------------------------------------------
__global__ void prep_weights(const float* __restrict__ W1, const float* __restrict__ W2,
                             __bf16* __restrict__ wA, __bf16* __restrict__ wB,
                             __bf16* __restrict__ w2) {
    int t = blockIdx.x * blockDim.x + threadIdx.x;
    if (t >= 3 * 32 * 32) return;
    int lane = t & 31;
    int frag = (t >> 5) & 31;
    int mat  = t >> 10;
    int nt = frag >> 2, ks = frag & 3;
    int n = nt * 16 + (lane & 15);
    int kbase = ks * 32 + ((lane < 16) ? 0 : 8);
    __bf16* dst = (mat == 0 ? wA : (mat == 1 ? wB : w2)) + (size_t)(frag * 32 + lane) * 16;
#pragma unroll
    for (int e = 0; e < 16; ++e) {
        int k = kbase + ((e < 8) ? e : e + 8);
        float v;
        if (mat == 0)      v = W1[k * DIM + n] - W1[(128 + k) * DIM + n];
        else if (mat == 1) v = W1[(128 + k) * DIM + n];
        else               v = W2[k * DIM + n];
        dst[e] = (__bf16)v;
    }
}

__global__ void zero_f32(float* __restrict__ p, size_t n) {
    size_t stride = (size_t)gridDim.x * blockDim.x;
    for (size_t i = blockIdx.x * (size_t)blockDim.x + threadIdx.x; i < n; i += stride)
        p[i] = 0.0f;
}

// ---------------------------------------------------------------------------
// Kernel 1: node precompute  P = feat@(W1a-W1b), Q = feat@W1b
// Weights staged in LDS via TDM; one wave = 16 rows; two passes (P, then Q)
// so only 8 accumulator tiles live (no spills).
// ---------------------------------------------------------------------------
__global__ __launch_bounds__(256) void node_gemm(
        const float* __restrict__ feat, const __bf16* __restrict__ wAB,
        float* __restrict__ P, float* __restrict__ Q, int N) {
    __shared__ __bf16 wlds[2 * 16384];    // wA ‖ wB fragment tables, 64KB
    __shared__ __bf16 alds[8][16 * DIM];  // 4KB per wave

    int tid  = threadIdx.x;
    int wave = tid >> 5, lane = tid & 31;

    stage_to_lds(wlds, wAB, 2 * 32768, tid);   // TDM + barrier

    int tile = blockIdx.x * 8 + wave;
    if (tile * 16 >= N) return;
    int row0 = tile * 16;

    __bf16* a_lds = alds[wave];
    const float* src = feat + (size_t)row0 * DIM;    // 16 consecutive rows
    for (int idx = lane; idx < 16 * DIM; idx += 32)
        a_lds[idx] = (__bf16)src[idx];
    asm volatile("s_wait_dscnt 0" ::: "memory");     // wave-private LDS

    int colb = lane & 15;
    int rhi  = (lane >> 4) * 8;

    for (int pass = 0; pass < 2; ++pass) {
        const __bf16* w = wlds + pass * 16384;
        float* dm = pass ? Q : P;
        f32x8 acc[8] = {};
#pragma unroll
        for (int ks = 0; ks < 4; ++ks) {
            bf16x16 a = load_a_frag_lds(a_lds, lane, ks);
            bf16x16 b[8];
#pragma unroll
            for (int nt = 0; nt < 8; ++nt)        // ds_load clause
                b[nt] = load_b_frag(w, nt, ks, lane);
#pragma unroll
            for (int nt = 0; nt < 8; ++nt)
                acc[nt] = __builtin_amdgcn_wmma_f32_16x16x32_bf16(
                    false, a, false, b[nt], (short)0, acc[nt], false, false);
        }
#pragma unroll
        for (int nt = 0; nt < 8; ++nt) {
            int col = nt * 16 + colb;
#pragma unroll
            for (int r = 0; r < 8; ++r)
                dm[(size_t)(row0 + rhi + r) * DIM + col] = acc[nt][r];
        }
    }
}

// ---------------------------------------------------------------------------
// Kernel 2: per-edge  h = GELU(P[src]+Q[dst]+rel@W1c+b1);  msg = h@W2+b2
//           atomic scatter-add into sums/counts (L2-resident, f32 atomics).
// W2 fragments staged in LDS via TDM; one wave = 16 edges.
// ---------------------------------------------------------------------------
__global__ __launch_bounds__(256) void edge_msg(
        const long long* __restrict__ ei, const float* __restrict__ coord,
        const float* __restrict__ P, const float* __restrict__ Q,
        const float* __restrict__ W1, const float* __restrict__ b1,
        const float* __restrict__ b2, const __bf16* __restrict__ w2,
        float* __restrict__ sums, float* __restrict__ counts, int E) {
    __shared__ __bf16 w2lds[16384];       // W2 fragment table, 32KB
    __shared__ __bf16 act[8][16 * DIM];   // 32KB
    __shared__ int    sdst[8][16];

    int tid  = threadIdx.x;
    int wave = tid >> 5, lane = tid & 31;

    stage_to_lds(w2lds, w2, 32768, tid);  // TDM + barrier

    int tile = blockIdx.x * 8 + wave;
    if (tile * 16 >= E) return;
    int e0 = tile * 16;

    if (lane < 16) sdst[wave][lane] = (int)ei[(size_t)E + e0 + lane];

    const float* w1c = W1 + 256 * DIM;   // rel_pos rows of W1 (3 x 128)
    for (int e = 0; e < 16; ++e) {
        int s = (int)ei[e0 + e];
        int d = (int)ei[(size_t)E + e0 + e];
        float rx = coord[d * 3 + 0] - coord[s * 3 + 0];
        float ry = coord[d * 3 + 1] - coord[s * 3 + 1];
        float rz = coord[d * 3 + 2] - coord[s * 3 + 2];
        const float* pr = P + (size_t)s * DIM;
        const float* qr = Q + (size_t)d * DIM;
#pragma unroll
        for (int i = 0; i < 4; ++i) {
            int j = lane + i * 32;
            float v = pr[j] + qr[j]
                    + rx * w1c[j] + ry * w1c[DIM + j] + rz * w1c[2 * DIM + j] + b1[j];
            float g = 0.5f * v * (1.0f + erff(v * 0.70710678118654752f));  // exact GELU
            act[wave][e * DIM + j] = (__bf16)g;
        }
    }
    asm volatile("s_wait_dscnt 0" ::: "memory");

    f32x8 acc[8] = {};
#pragma unroll
    for (int ks = 0; ks < 4; ++ks) {
        bf16x16 a = load_a_frag_lds(act[wave], lane, ks);
        bf16x16 b[8];
#pragma unroll
        for (int nt = 0; nt < 8; ++nt)
            b[nt] = load_b_frag(w2lds, nt, ks, lane);
#pragma unroll
        for (int nt = 0; nt < 8; ++nt)
            acc[nt] = __builtin_amdgcn_wmma_f32_16x16x32_bf16(
                false, a, false, b[nt], (short)0, acc[nt], false, false);
    }

    int colb = lane & 15;
    int rhi  = (lane >> 4) * 8;
#pragma unroll
    for (int nt = 0; nt < 8; ++nt) {
        int col = nt * 16 + colb;
        float bb = b2[col];
#pragma unroll
        for (int r = 0; r < 8; ++r) {
            int d = sdst[wave][rhi + r];
            atomicAdd(&sums[(size_t)d * DIM + col], acc[nt][r] + bb);
        }
    }
    if (lane < 16)
        atomicAdd(&counts[sdst[wave][lane]], 1.0f);
}

// ---------------------------------------------------------------------------
// Kernel 3: x = sums/max(cnt,1) + feat; LayerNorm(x)*gamma + beta
// ---------------------------------------------------------------------------
__global__ void finalize(const float* __restrict__ sums, const float* __restrict__ counts,
                         const float* __restrict__ feat, const float* __restrict__ gamma,
                         const float* __restrict__ beta, float* __restrict__ out, int N) {
    int wave = threadIdx.x >> 5, lane = threadIdx.x & 31;
    int row = blockIdx.x * 8 + wave;
    if (row >= N) return;
    float inv = 1.0f / fmaxf(counts[row], 1.0f);
    const float* sr = sums + (size_t)row * DIM;
    const float* fr = feat + (size_t)row * DIM;
    float x[4], s = 0.0f;
#pragma unroll
    for (int i = 0; i < 4; ++i) {
        int j = lane + i * 32;
        x[i] = sr[j] * inv + fr[j];
        s += x[i];
    }
#pragma unroll
    for (int m = 16; m >= 1; m >>= 1) s += __shfl_xor(s, m, 32);
    float mu = s * (1.0f / DIM);
    float v = 0.0f;
#pragma unroll
    for (int i = 0; i < 4; ++i) { float d = x[i] - mu; v += d * d; }
#pragma unroll
    for (int m = 16; m >= 1; m >>= 1) v += __shfl_xor(v, m, 32);
    float r = rsqrtf(v * (1.0f / DIM) + 1e-5f);
    float* orow = out + (size_t)row * DIM;
#pragma unroll
    for (int i = 0; i < 4; ++i) {
        int j = lane + i * 32;
        orow[j] = (x[i] - mu) * r * gamma[j] + beta[j];
    }
}

// ---------------------------------------------------------------------------
extern "C" void kernel_launch(void* const* d_in, const int* in_sizes, int n_in,
                              void* d_out, int out_size, void* d_ws, size_t ws_size,
                              hipStream_t stream) {
    const float*     feat  = (const float*)d_in[0];
    const long long* ei    = (const long long*)d_in[1];
    const float*     coord = (const float*)d_in[2];
    const float*     W1    = (const float*)d_in[3];
    const float*     b1    = (const float*)d_in[4];
    const float*     W2    = (const float*)d_in[5];
    const float*     b2    = (const float*)d_in[6];
    const float*     gamma = (const float*)d_in[7];
    const float*     beta  = (const float*)d_in[8];
    float* out = (float*)d_out;

    int N = in_sizes[0] / DIM;
    int E = in_sizes[1] / 2;

    // workspace layout
    char* ws = (char*)d_ws;
    __bf16* wA = (__bf16*)ws;                  // 32KB  (B-fragment swizzled) — wA‖wB contiguous
    __bf16* wB = (__bf16*)(ws + 32768);        // 32KB
    __bf16* w2 = (__bf16*)(ws + 65536);        // 32KB
    float*  P  = (float*)(ws + 98304);         // N*128 f32
    float*  Q  = P + (size_t)N * DIM;          // N*128 f32
    float*  sums   = Q + (size_t)N * DIM;      // N*128 f32
    float*  counts = sums + (size_t)N * DIM;   // N f32

    prep_weights<<<12, 256, 0, stream>>>(W1, W2, wA, wB, w2);
    zero_f32<<<1024, 256, 0, stream>>>(sums, (size_t)N * DIM + (size_t)N);

    int ntiles = (N + 15) / 16;
    node_gemm<<<(ntiles + 7) / 8, 256, 0, stream>>>(feat, wA, P, Q, N);

    int etiles = (E + 15) / 16;
    edge_msg<<<(etiles + 7) / 8, 256, 0, stream>>>(ei, coord, P, Q, W1, b1, b2, w2,
                                                   sums, counts, E);

    finalize<<<(N + 7) / 8, 256, 0, stream>>>(sums, counts, feat, gamma, beta, out, N);
}
